// SVDQuantLinear_25202868093527
// MI455X (gfx1250) — compile-verified, matrix operand order checked by reading, not run on hardware
//
#include <hip/hip_runtime.h>
#include <hip/hip_bf16.h>

#define M_DIM 8192
#define N_DIM 4096
#define K_DIM 4096
#define BM 256
#define BN 128
#define BK 32
#define LDS_ROW_U4 5  // 80 B per LDS row: 64 B data + 16 B pad -> conflict-free frag reads

typedef __attribute__((ext_vector_type(16))) __bf16 v16bf;
typedef __attribute__((ext_vector_type(8)))  float  v8f;
typedef int v4i __attribute__((vector_size(16)));

union FragU { v16bf v; uint4 u[2]; };

__device__ __forceinline__ unsigned short f2bf(float f) {
  __hip_bfloat16 h = __float2bfloat16(f);
  return *reinterpret_cast<unsigned short*>(&h);
}

// ---------------- async global -> LDS (CDNA5 ASYNCcnt path) -----------------
#if __has_builtin(__builtin_amdgcn_global_load_async_to_lds_b128)
#define HAVE_ASYNC_LDS 1
__device__ __forceinline__ void async_copy16(const void* gp, void* lp) {
  __builtin_amdgcn_global_load_async_to_lds_b128(
      (v4i*)gp, (v4i*)lp, /*offset=*/0, /*cpol=*/0);
}
__device__ __forceinline__ void async_wait0() {
#if __has_builtin(__builtin_amdgcn_s_wait_asynccnt)
  __builtin_amdgcn_s_wait_asynccnt(0);
#else
  asm volatile("s_wait_asynccnt 0" ::: "memory");
#endif
}
#else
#define HAVE_ASYNC_LDS 0
__device__ __forceinline__ void async_copy16(const void* gp, void* lp) {
  *reinterpret_cast<uint4*>(lp) = *reinterpret_cast<const uint4*>(gp);
}
__device__ __forceinline__ void async_wait0() {}
#endif

// ---- x fp32 -> bf16, vectorized 4/thread ------------------------------------
__global__ __launch_bounds__(256) void xcast_kernel(const float* __restrict__ x,
                                                    unsigned short* __restrict__ xb) {
  size_t i = ((size_t)blockIdx.x * 256u + threadIdx.x) * 4u;
  float4 v = *reinterpret_cast<const float4*>(x + i);
  ushort4 o;
  o.x = f2bf(v.x); o.y = f2bf(v.y); o.z = f2bf(v.z); o.w = f2bf(v.w);
  *reinterpret_cast<ushort4*>(xb + i) = o;
}

// ---- W_eff[n,k] = w_q[n,k]*scale[n] + sum_r lora_B[n,r]*S[r]*lora_A[r,k] ----
__global__ __launch_bounds__(256) void wprep_kernel(
    const int* __restrict__ wq, const float* __restrict__ wscale,
    const float* __restrict__ lA, const float* __restrict__ lS,
    const float* __restrict__ lB, unsigned short* __restrict__ wb) {
  __shared__ float coef[32];
  const int n = blockIdx.x;
  const int tid = threadIdx.x;
  if (tid < 32) coef[tid] = lB[(size_t)n * 32 + tid] * lS[tid];
  __syncthreads();
  const float ws = wscale[n];
  const size_t rowoff = (size_t)n * K_DIM;
  for (int k = tid; k < K_DIM; k += 256) {
    float acc = (float)wq[rowoff + k] * ws;
#pragma unroll
    for (int r = 0; r < 32; ++r) acc += coef[r] * lA[(size_t)r * K_DIM + k];
    wb[rowoff + k] = f2bf(acc);
  }
}

// ---- load one 16x32 bf16 fragment (A-style layout; B uses same since NT) ----
__device__ __forceinline__ v16bf frag_ld(const uint4* tile, int row0, int lane) {
  const int l  = lane & 15;
  const int hi = (lane >> 4) & 1;
  const unsigned char* p =
      reinterpret_cast<const unsigned char*>(tile) + (size_t)(row0 + l) * 80 + hi * 16;
  FragU f;
  f.u[0] = *reinterpret_cast<const uint4*>(p);        // K 0..7   (or 8..15)
  f.u[1] = *reinterpret_cast<const uint4*>(p + 32);   // K 16..23 (or 24..31)
  return f.v;
}

// ---- issue one k-step's tile copies (X: 4 chunks/thread, W: 2) -------------
__device__ __forceinline__ void issue_tiles(const unsigned short* __restrict__ Xb,
                                            const unsigned short* __restrict__ Wb,
                                            uint4* bufX, uint4* bufW,
                                            int m0, int n0, int k0, int tid) {
#pragma unroll
  for (int t = 0; t < 4; ++t) {
    int c = tid + t * 256, row = c >> 2, off = c & 3;  // 1024 chunks for X
    async_copy16(Xb + (size_t)(m0 + row) * K_DIM + k0 + off * 8,
                 bufX + row * LDS_ROW_U4 + off);
  }
#pragma unroll
  for (int t = 0; t < 2; ++t) {
    int c = tid + t * 256, row = c >> 2, off = c & 3;  // 512 chunks for W
    async_copy16(Wb + (size_t)(n0 + row) * K_DIM + k0 + off * 8,
                 bufW + row * LDS_ROW_U4 + off);
  }
}

// ---- main GEMM: out = Xb @ Wb^T + bias, bf16 WMMA, f32 accumulate ----------
__global__ __launch_bounds__(256) void gemm_kernel(
    const unsigned short* __restrict__ Xb, const unsigned short* __restrict__ Wb,
    const float* __restrict__ bias, float* __restrict__ out) {
  __shared__ uint4 ldsX[2][BM * LDS_ROW_U4];  // 2 x 20 KB
  __shared__ uint4 ldsW[2][BN * LDS_ROW_U4];  // 2 x 10 KB

  const int tid  = threadIdx.x;
  const int lane = tid & 31;
  const int wave = tid >> 5;
  const int wm   = wave >> 1;  // 0..3 -> 64-row M strip
  const int wn   = wave & 1;   // 0..1 -> 64-col N strip
  const int m0   = blockIdx.y * BM;
  const int n0   = blockIdx.x * BN;

  v8f acc[4][4];
#pragma unroll
  for (int mi = 0; mi < 4; ++mi)
#pragma unroll
    for (int ni = 0; ni < 4; ++ni) acc[mi][ni] = 0;

  // prologue: stage tile 0
  issue_tiles(Xb, Wb, ldsX[0], ldsW[0], m0, n0, 0, tid);
  async_wait0();
  __syncthreads();

  int cur = 0;
  for (int k0 = 0; k0 < K_DIM; k0 += BK) {
    const int nxt = cur ^ 1;
    if (k0 + BK < K_DIM)  // stream next tile while computing on current
      issue_tiles(Xb, Wb, ldsX[nxt], ldsW[nxt], m0, n0, k0 + BK, tid);

    v16bf afrag[4], bfrag[4];
#pragma unroll
    for (int mi = 0; mi < 4; ++mi)
      afrag[mi] = frag_ld(ldsX[cur], wm * 64 + mi * 16, lane);
#pragma unroll
    for (int ni = 0; ni < 4; ++ni)
      bfrag[ni] = frag_ld(ldsW[cur], wn * 64 + ni * 16, lane);

#pragma unroll
    for (int mi = 0; mi < 4; ++mi)
#pragma unroll
      for (int ni = 0; ni < 4; ++ni)
        acc[mi][ni] = __builtin_amdgcn_wmma_f32_16x16x32_bf16(
            false, afrag[mi], false, bfrag[ni], (short)0, acc[mi][ni], false, false);

    async_wait0();     // next tile fully landed in LDS
    __syncthreads();   // all waves done reading cur / writing nxt
    cur = nxt;
  }

  // epilogue: C layout -> VGPR r holds (m = r + 8*hi, n = lane&15)
  const int l  = lane & 15;
  const int hi = (lane >> 4) & 1;
#pragma unroll
  for (int mi = 0; mi < 4; ++mi) {
#pragma unroll
    for (int ni = 0; ni < 4; ++ni) {
      const int gn = n0 + wn * 64 + ni * 16 + l;
      const float bv = bias[gn];
      const size_t base = (size_t)(m0 + wm * 64 + mi * 16 + hi * 8) * N_DIM + gn;
#pragma unroll
      for (int r = 0; r < 8; ++r)
        out[base + (size_t)r * N_DIM] = acc[mi][ni][r] + bv;
    }
  }
}

extern "C" void kernel_launch(void* const* d_in, const int* in_sizes, int n_in,
                              void* d_out, int out_size, void* d_ws, size_t ws_size,
                              hipStream_t stream) {
  (void)in_sizes; (void)n_in; (void)out_size; (void)ws_size;
  const float* x       = (const float*)d_in[0];
  const int*   w_q     = (const int*)d_in[1];
  const float* w_scale = (const float*)d_in[2];
  const float* bias    = (const float*)d_in[3];
  const float* lora_A  = (const float*)d_in[4];
  const float* lora_S  = (const float*)d_in[5];
  const float* lora_B  = (const float*)d_in[6];
  float* out = (float*)d_out;

  // workspace: [0, 32MB) = W_eff bf16 ; [32MB, 96MB) = x bf16
  unsigned short* Wb = (unsigned short*)d_ws;
  unsigned short* Xb = (unsigned short*)((char*)d_ws + (size_t)N_DIM * K_DIM * 2);

  xcast_kernel<<<(M_DIM * (size_t)K_DIM) / (256 * 4), 256, 0, stream>>>(x, Xb);
  wprep_kernel<<<N_DIM, 256, 0, stream>>>(w_q, w_scale, lora_A, lora_S, lora_B, Wb);
  dim3 grid(N_DIM / BN, M_DIM / BM);  // 32 x 32
  gemm_kernel<<<grid, 256, 0, stream>>>(Xb, Wb, bias, out);
}
